// KimiMLAAttention_68195490726077
// MI455X (gfx1250) — matrix operand order repaired and checked by reading, not compiled, not run on hardware
//
#include <hip/hip_runtime.h>

typedef __attribute__((ext_vector_type(16))) __bf16        v16bf;
typedef __attribute__((ext_vector_type(8)))  float         v8f;
typedef __attribute__((ext_vector_type(8)))  unsigned int  v8u;

// ---------------------------------------------------------------- helpers ---
__device__ __forceinline__ unsigned short f2bf(float x) {
  unsigned int u = __builtin_bit_cast(unsigned int, x);
  u = (u + 0x7FFFu + ((u >> 16) & 1u)) >> 16;   // round-to-nearest-even
  return (unsigned short)u;
}

// ---- async global->LDS 16-byte copy (ASYNCcnt path), with safe fallback ----
#if defined(__has_builtin)
#if __has_builtin(__builtin_amdgcn_global_load_async_to_lds_b128)
#define HAVE_ASYNC_LDS 1
#endif
#if __has_builtin(__builtin_amdgcn_s_wait_asynccnt)
#define HAVE_WAIT_ASYNC 1
#endif
#endif

typedef int vi4 __attribute__((vector_size(16)));
typedef __attribute__((address_space(1))) vi4 gvi4;   // global int4
typedef __attribute__((address_space(3))) vi4 lvi4;   // LDS int4

__device__ __forceinline__ void async_copy16(const unsigned short* g, unsigned short* l) {
#if defined(HAVE_ASYNC_LDS)
  __builtin_amdgcn_global_load_async_to_lds_b128(
      (gvi4*)(unsigned long long)(uintptr_t)g,
      (lvi4*)(unsigned int)(uintptr_t)l,
      0, 0);
#else
  *(uint4*)l = *(const uint4*)g;
#endif
}
__device__ __forceinline__ void async_wait_all() {
#if defined(HAVE_ASYNC_LDS)
#if defined(HAVE_WAIT_ASYNC)
  __builtin_amdgcn_s_wait_asynccnt(0);
#else
  asm volatile("s_wait_asynccnt 0x0" ::: "memory");
#endif
#endif
}

// A-fragment (16x32 bf16, M x K): lane half h holds K = (j/4)*16 + 8h + 2*(j%4)
template <int STRIDE>
__device__ __forceinline__ v16bf load_fragA(const unsigned short* base, int row_base,
                                            int kbase, int lane) {
  const int m = row_base + (lane & 15);
  const int h = lane >> 4;
  const unsigned int* p = (const unsigned int*)(base + (size_t)m * STRIDE);
  v8u u;
#pragma unroll
  for (int j = 0; j < 8; ++j) {
    int k0 = kbase + ((j >> 2) << 4) + (h << 3) + ((j & 3) << 1);
    u[j] = p[k0 >> 1];
  }
  return __builtin_bit_cast(v16bf, u);
}

// B-fragment (32x16 bf16, K x N), LDS stored as [n][k]: lane half h holds K = (j/4)*16 + 8h + 2*(j%4)
template <int STRIDE>
__device__ __forceinline__ v16bf load_fragB(const unsigned short* base, int nbase,
                                            int kbase, int lane) {
  const int n = nbase + (lane & 15);
  const int h = lane >> 4;
  const unsigned int* p = (const unsigned int*)(base + (size_t)n * STRIDE);
  v8u u;
#pragma unroll
  for (int j = 0; j < 8; ++j) {
    int k0 = kbase + ((j >> 2) << 4) + (h << 3) + ((j & 3) << 1);
    u[j] = p[k0 >> 1];
  }
  return __builtin_bit_cast(v16bf, u);
}

#define WMMA_BF16(a, b, c) \
  __builtin_amdgcn_wmma_f32_16x16x32_bf16(false, (a), false, (b), (short)0, (c), false, false)

// -------------------------------------------------- fp32 -> bf16 convert ---
__global__ __launch_bounds__(256) void convert_bf16(const float* __restrict__ in,
                                                    unsigned short* __restrict__ out,
                                                    int count) {
  int base = (blockIdx.x * 256 + threadIdx.x) * 4;
  if (base + 3 < count) {
#pragma unroll
    for (int i = 0; i < 4; ++i) out[base + i] = f2bf(in[base + i]);
  }
}

// --------------------------- fp32 [K][N] -> bf16 [N][K] tiled transpose ----
__global__ __launch_bounds__(256) void transpose_bf16(const float* __restrict__ in,
                                                      unsigned short* __restrict__ out,
                                                      int K, int N) {
  __shared__ float t[32][33];
  const int k0 = blockIdx.y * 32, n0 = blockIdx.x * 32;
  const int tx = threadIdx.x & 31, ty = threadIdx.x >> 5;  // 32 x 8
#pragma unroll
  for (int i = 0; i < 4; ++i) t[ty + 8 * i][tx] = in[(size_t)(k0 + ty + 8 * i) * N + n0 + tx];
  __syncthreads();
#pragma unroll
  for (int i = 0; i < 4; ++i)
    out[(size_t)(n0 + ty + 8 * i) * K + k0 + tx] = f2bf(t[tx][ty + 8 * i]);
}

// ------------------------------------------------------------- tiled GEMM ---
// C[M,N] = A[M,K] @ Bt[N,K]^T : bf16 operands, fp32 accumulate/output.
// M % 128 == 0, K % 32 == 0; N guarded on store (Bt rows padded by caller).
__global__ __launch_bounds__(256) void gemm_bf16(const unsigned short* __restrict__ A,
                                                 const unsigned short* __restrict__ Bt,
                                                 float* __restrict__ C,
                                                 int M, int N, int K) {
  __shared__ unsigned short As[128][40];  // [m][k], 80B rows (16B aligned)
  __shared__ unsigned short Bs[128][40];  // [n][k]
  const int tid = threadIdx.x, lane = tid & 31, wid = tid >> 5;
  const int wm = wid & 3, wn = wid >> 2;  // 4x2 wave grid
  const int m0 = blockIdx.y * 128, n0 = blockIdx.x * 128;

  v8f acc[2][4] = {};
  for (int k0 = 0; k0 < K; k0 += 32) {
    __syncthreads();
#pragma unroll
    for (int it = 0; it < 2; ++it) {      // A tile: 128 rows x 64B
      int lin = tid + it * 256;
      int r = lin >> 2, c = (lin & 3) * 8;
      async_copy16(A + (size_t)(m0 + r) * K + k0 + c, &As[r][c]);
    }
#pragma unroll
    for (int it = 0; it < 2; ++it) {      // B tile: 128 rows x 64B
      int lin = tid + it * 256;
      int r = lin >> 2, c = (lin & 3) * 8;
      async_copy16(Bt + (size_t)(n0 + r) * K + k0 + c, &Bs[r][c]);
    }
    async_wait_all();
    __syncthreads();
    v16bf aF[2];
#pragma unroll
    for (int i = 0; i < 2; ++i) aF[i] = load_fragA<40>(&As[0][0], wm * 32 + i * 16, 0, lane);
#pragma unroll
    for (int j = 0; j < 4; ++j) {
      v16bf bF = load_fragB<40>(&Bs[0][0], wn * 64 + j * 16, 0, lane);
#pragma unroll
      for (int i = 0; i < 2; ++i) acc[i][j] = WMMA_BF16(aF[i], bF, acc[i][j]);
    }
  }
#pragma unroll
  for (int i = 0; i < 2; ++i)
#pragma unroll
    for (int j = 0; j < 4; ++j)
#pragma unroll
      for (int r = 0; r < 8; ++r) {
        int m = m0 + wm * 32 + i * 16 + r + 8 * (lane >> 4);
        int n = n0 + wn * 64 + j * 16 + (lane & 15);
        if (n < N) C[(size_t)m * N + n] = acc[i][j][r];
      }
}

// --------------------------------------------------- RMS-norm + K-rot RoPE ---
__global__ __launch_bounds__(128) void rmsnorm_rope(const float* __restrict__ ckv,
                                                    const float* __restrict__ lnw,
                                                    unsigned short* __restrict__ cnormb,
                                                    float* __restrict__ krot) {
  const int s = blockIdx.x, tid = threadIdx.x, lane = tid & 31, wid = tid >> 5;
  __shared__ float red[4];
  const float* row = ckv + (size_t)s * 576;
  float ss = 0.f;
  for (int c = tid; c < 512; c += 128) { float x = row[c]; ss += x * x; }
#pragma unroll
  for (int off = 16; off >= 1; off >>= 1) ss += __shfl_xor(ss, off, 32);
  if (lane == 0) red[wid] = ss;
  __syncthreads();
  float rstd = rsqrtf((red[0] + red[1] + red[2] + red[3]) * (1.0f / 512.0f) + 1e-6f);
  for (int c = tid; c < 512; c += 128)
    cnormb[(size_t)s * 512 + c] = f2bf(lnw[c] * row[c] * rstd);
  if (tid < 32) {
    int i = tid;
    float freq = __powf(10000.0f, -(float)(2 * i) / 64.0f);
    float ang = (float)s * freq;
    float cs = cosf(ang), sn = sinf(ang);
    float x1 = row[512 + i], x2 = row[544 + i];
    krot[(size_t)s * 64 + i]      = x1 * cs - x2 * sn;
    krot[(size_t)s * 64 + 32 + i] = x2 * cs + x1 * sn;
  }
}

// -------------------- Q RoPE+scale; per-head bf16 Q, K, V(transposed) ------
__global__ __launch_bounds__(256) void prep_qkv(const float* __restrict__ qg,
                                                const float* __restrict__ kvf,
                                                const float* __restrict__ krot,
                                                unsigned short* __restrict__ Qb,
                                                unsigned short* __restrict__ Kb,
                                                unsigned short* __restrict__ Vt) {
  const int s = blockIdx.x, tid = threadIdx.x;
  const float scale = 0.07216878364870323f;  // 192^-0.5 folded into Q
  for (int idx = tid; idx < 4096; idx += 256) {            // Q pass part
    int h = idx >> 7, d = idx & 127;
    Qb[((size_t)h * 2048 + s) * 192 + d] = f2bf(qg[(size_t)s * 6144 + h * 192 + d] * scale);
  }
  for (int idx = tid; idx < 1024; idx += 256) {            // Q rope part
    int h = idx >> 5, i = idx & 31;
    float freq = __powf(10000.0f, -(float)(2 * i) / 64.0f);
    float ang = (float)s * freq;
    float cs = cosf(ang), sn = sinf(ang);
    float x1 = qg[(size_t)s * 6144 + h * 192 + 128 + i];
    float x2 = qg[(size_t)s * 6144 + h * 192 + 160 + i];
    Qb[((size_t)h * 2048 + s) * 192 + 128 + i] = f2bf((x1 * cs - x2 * sn) * scale);
    Qb[((size_t)h * 2048 + s) * 192 + 160 + i] = f2bf((x2 * cs + x1 * sn) * scale);
  }
  for (int idx = tid; idx < 8192; idx += 256) {            // K pass + V (V transposed)
    int h = idx >> 8, d = idx & 255;
    float v = kvf[(size_t)s * 8192 + h * 256 + d];
    if (d < 128) Kb[((size_t)h * 2048 + s) * 192 + d] = f2bf(v);
    else         Vt[((size_t)h * 128 + (d - 128)) * 2048 + s] = f2bf(v);
  }
  for (int idx = tid; idx < 2048; idx += 256) {            // K rot broadcast
    int h = idx >> 6, i = idx & 63;
    Kb[((size_t)h * 2048 + s) * 192 + 128 + i] = f2bf(krot[(size_t)s * 64 + i]);
  }
}

// --------------------------------------------------------- flash attention ---
// grid (16 q-blocks of 128 rows, 32 heads), block 256 (8 waves, 16 rows/wave).
__global__ __launch_bounds__(256) void flash_mla(const unsigned short* __restrict__ Qb,
                                                 const unsigned short* __restrict__ Kb,
                                                 const unsigned short* __restrict__ Vt,
                                                 unsigned short* __restrict__ attnb) {
  __shared__ unsigned short Ks[64][200];      // K tile [key][192], 400B rows
  __shared__ unsigned short Vts[128][72];     // V tile [d][key], 144B rows
  __shared__ unsigned short Ps[8][16][68];    // per-wave P bounce buffer
  const int tid = threadIdx.x, lane = tid & 31, wid = tid >> 5;
  const int qb = blockIdx.x, h = blockIdx.y;

  // stage Q (128 rows) through Ks in two 64-row rounds; lift to VGPR fragments
  v16bf qf[6];
  for (int half = 0; half < 2; ++half) {
    __syncthreads();
    const unsigned short* qsrc = Qb + ((size_t)h * 2048 + qb * 128 + half * 64) * 192;
    for (int idx = tid; idx < 1536; idx += 256) {
      int r = idx / 24, c = (idx % 24) * 8;
      async_copy16(qsrc + (size_t)r * 192 + c, &Ks[r][c]);
    }
    async_wait_all();
    __syncthreads();
    if ((wid >> 2) == half) {
#pragma unroll
      for (int kt = 0; kt < 6; ++kt)
        qf[kt] = load_fragA<200>(&Ks[0][0], (wid & 3) * 16, kt * 32, lane);
    }
  }

  float mrow[8], lrow[8];
#pragma unroll
  for (int r = 0; r < 8; ++r) { mrow[r] = -3.0e38f; lrow[r] = 0.f; }
  v8f o[8] = {};

  const int jbmax = 2 * qb + 1;
  for (int jb = 0; jb <= jbmax; ++jb) {
    __syncthreads();
    {
      const unsigned short* ksrc = Kb + ((size_t)h * 2048 + jb * 64) * 192;
      for (int idx = tid; idx < 1536; idx += 256) {
        int r = idx / 24, c = (idx % 24) * 8;
        async_copy16(ksrc + (size_t)r * 192 + c, &Ks[r][c]);
      }
      const unsigned short* vsrc = Vt + (size_t)h * 128 * 2048 + jb * 64;
      for (int idx = tid; idx < 1024; idx += 256) {
        int d = idx >> 3, c = (idx & 7) * 8;
        async_copy16(vsrc + (size_t)d * 2048 + c, &Vts[d][c]);
      }
    }
    async_wait_all();
    __syncthreads();

    v8f s4[4] = {};
#pragma unroll
    for (int kt = 0; kt < 6; ++kt)
#pragma unroll
      for (int t = 0; t < 4; ++t) {
        v16bf bK = load_fragB<200>(&Ks[0][0], t * 16, kt * 32, lane);
        s4[t] = WMMA_BF16(qf[kt], bK, s4[t]);
      }

    const bool diag = (jb >= 2 * qb);
    const int rg = qb * 128 + wid * 16 + 8 * (lane >> 4);
    const int cg = jb * 64 + (lane & 15);
#pragma unroll
    for (int r = 0; r < 8; ++r) {
      int row = rg + r;
      float mx = -3.0e38f;
#pragma unroll
      for (int t = 0; t < 4; ++t) {
        float v = s4[t][r];
        if (diag && (cg + t * 16) > row) { v = -3.0e38f; s4[t][r] = v; }
        mx = fmaxf(mx, v);
      }
#pragma unroll
      for (int off = 8; off >= 1; off >>= 1) mx = fmaxf(mx, __shfl_xor(mx, off, 32));
      float newm = fmaxf(mrow[r], mx);
      float alpha = __expf(mrow[r] - newm);
      float sum = 0.f;
#pragma unroll
      for (int t = 0; t < 4; ++t) {
        float p = __expf(s4[t][r] - newm);
        sum += p;
        Ps[wid][r + 8 * (lane >> 4)][t * 16 + (lane & 15)] = f2bf(p);
      }
#pragma unroll
      for (int off = 8; off >= 1; off >>= 1) sum += __shfl_xor(sum, off, 32);
      mrow[r] = newm;
      lrow[r] = lrow[r] * alpha + sum;
#pragma unroll
      for (int t = 0; t < 8; ++t) o[t][r] *= alpha;
    }

    // P(16x64) @ V(64x128): per-wave, LDS in-order within wave covers P RAW
#pragma unroll
    for (int kt = 0; kt < 2; ++kt) {
      v16bf aP = load_fragA<68>(&Ps[wid][0][0], 0, kt * 32, lane);
#pragma unroll
      for (int t = 0; t < 8; ++t) {
        v16bf bV = load_fragB<72>(&Vts[0][0], t * 16, kt * 32, lane);
        o[t] = WMMA_BF16(aP, bV, o[t]);
      }
    }
  }

#pragma unroll
  for (int r = 0; r < 8; ++r) {
    int row = qb * 128 + wid * 16 + 8 * (lane >> 4) + r;
    float inv = 1.0f / lrow[r];
#pragma unroll
    for (int t = 0; t < 8; ++t)
      attnb[(size_t)row * 4096 + h * 128 + t * 16 + (lane & 15)] = f2bf(o[t][r] * inv);
  }
}

// ---------------------------------------------------------------- launcher ---
#define ALIGN256(x) (((x) + 255) & ~(size_t)255)
static constexpr size_t SZ_HIDB  = (size_t)2048 * 4096 * 2;
static constexpr size_t SZ_WQT   = (size_t)6144 * 4096 * 2;
static constexpr size_t SZ_WKAT  = (size_t)640  * 4096 * 2;   // 576 rows + OOB pad
static constexpr size_t SZ_WKBT  = (size_t)8192 * 512  * 2;
static constexpr size_t SZ_WOT   = (size_t)4096 * 4096 * 2;
static constexpr size_t SZ_QG    = (size_t)2048 * 6144 * 4;
static constexpr size_t SZ_CKV   = (size_t)2048 * 576  * 4;
static constexpr size_t SZ_CNB   = (size_t)2048 * 512  * 2;
static constexpr size_t SZ_KROT  = (size_t)2048 * 64   * 4;
static constexpr size_t SZ_KVF   = (size_t)2048 * 8192 * 4;
static constexpr size_t SZ_QB    = (size_t)32 * 2048 * 192 * 2;
static constexpr size_t SZ_KB    = (size_t)32 * 2048 * 192 * 2;
static constexpr size_t SZ_VT    = (size_t)32 * 128 * 2048 * 2;
static constexpr size_t SZ_ATNB  = (size_t)2048 * 4096 * 2;

static constexpr size_t OFF_HIDB = 0;
static constexpr size_t OFF_WQT  = OFF_HIDB + ALIGN256(SZ_HIDB);
static constexpr size_t OFF_WKAT = OFF_WQT  + ALIGN256(SZ_WQT);
static constexpr size_t OFF_WKBT = OFF_WKAT + ALIGN256(SZ_WKAT);
static constexpr size_t OFF_WOT  = OFF_WKBT + ALIGN256(SZ_WKBT);
static constexpr size_t OFF_QG   = OFF_WOT  + ALIGN256(SZ_WOT);
static constexpr size_t OFF_CKV  = OFF_QG   + ALIGN256(SZ_QG);
static constexpr size_t OFF_CNB  = OFF_CKV  + ALIGN256(SZ_CKV);
static constexpr size_t OFF_KROT = OFF_CNB  + ALIGN256(SZ_CNB);
static constexpr size_t OFF_KVF  = OFF_KROT + ALIGN256(SZ_KROT);
static constexpr size_t OFF_QB   = OFF_KVF  + ALIGN256(SZ_KVF);
static constexpr size_t OFF_KB   = OFF_QB   + ALIGN256(SZ_QB);
static constexpr size_t OFF_VT   = OFF_KB   + ALIGN256(SZ_KB);
static constexpr size_t OFF_ATNB = OFF_VT   + ALIGN256(SZ_VT);

extern "C" void kernel_launch(void* const* d_in, const int* in_sizes, int n_in,
                              void* d_out, int out_size, void* d_ws, size_t ws_size,
                              hipStream_t stream) {
  (void)in_sizes; (void)n_in; (void)out_size; (void)ws_size;
  const float* hidden = (const float*)d_in[1];
  const float* Wq     = (const float*)d_in[2];
  const float* Wkv_a  = (const float*)d_in[3];
  const float* lnw    = (const float*)d_in[4];
  const float* Wkv_b  = (const float*)d_in[5];
  const float* Wo     = (const float*)d_in[6];

  char* ws = (char*)d_ws;
  unsigned short* hidb  = (unsigned short*)(ws + OFF_HIDB);
  unsigned short* Wq_t  = (unsigned short*)(ws + OFF_WQT);
  unsigned short* Wka_t = (unsigned short*)(ws + OFF_WKAT);
  unsigned short* Wkb_t = (unsigned short*)(ws + OFF_WKBT);
  unsigned short* Wo_t  = (unsigned short*)(ws + OFF_WOT);
  float*          qg    = (float*)(ws + OFF_QG);
  float*          ckv   = (float*)(ws + OFF_CKV);
  unsigned short* cnb   = (unsigned short*)(ws + OFF_CNB);
  float*          krot  = (float*)(ws + OFF_KROT);
  float*          kvf   = (float*)(ws + OFF_KVF);
  unsigned short* Qb    = (unsigned short*)(ws + OFF_QB);
  unsigned short* Kb    = (unsigned short*)(ws + OFF_KB);
  unsigned short* Vt    = (unsigned short*)(ws + OFF_VT);
  unsigned short* atnb  = (unsigned short*)(ws + OFF_ATNB);
  float*          out   = (float*)d_out;

  // one-time bf16 conversion / weight transposition
  convert_bf16<<<dim3(2048 * 4096 / 1024), 256, 0, stream>>>(hidden, hidb, 2048 * 4096);
  transpose_bf16<<<dim3(192, 128), 256, 0, stream>>>(Wq,    Wq_t,  4096, 6144);
  transpose_bf16<<<dim3(18, 128),  256, 0, stream>>>(Wkv_a, Wka_t, 4096, 576);
  transpose_bf16<<<dim3(256, 16),  256, 0, stream>>>(Wkv_b, Wkb_t, 512,  8192);
  transpose_bf16<<<dim3(128, 128), 256, 0, stream>>>(Wo,    Wo_t,  4096, 4096);

  gemm_bf16<<<dim3(48, 16), 256, 0, stream>>>(hidb, Wq_t,  qg,  2048, 6144, 4096);
  gemm_bf16<<<dim3(5, 16),  256, 0, stream>>>(hidb, Wka_t, ckv, 2048, 576,  4096);
  rmsnorm_rope<<<2048, 128, 0, stream>>>(ckv, lnw, cnb, krot);
  gemm_bf16<<<dim3(64, 16), 256, 0, stream>>>(cnb, Wkb_t, kvf, 2048, 8192, 512);
  prep_qkv<<<2048, 256, 0, stream>>>(qg, kvf, krot, Qb, Kb, Vt);
  flash_mla<<<dim3(16, 32), 256, 0, stream>>>(Qb, Kb, Vt, atnb);
  gemm_bf16<<<dim3(32, 16), 256, 0, stream>>>(atnb, Wo_t, out, 2048, 4096, 4096);
}
